// LSTMDecoder_54030688583991
// MI455X (gfx1250) — matrix operand order repaired and checked by reading, not compiled, not run on hardware
//
#include <hip/hip_runtime.h>
#include <hip/hip_bf16.h>

// ---------------------------------------------------------------------------
// Types for CDNA5 WMMA (wave32): D[16x16 f32] = A[16x32 bf16] * B[32x16 bf16] + C
// ---------------------------------------------------------------------------
typedef __attribute__((ext_vector_type(16))) __bf16 v16bf;
typedef __attribute__((ext_vector_type(8)))  float  v8f;
typedef __attribute__((ext_vector_type(4)))  int    v4i;

#define B_SZ 32
#define T_SZ 512
#define E_SZ 768
#define H_SZ 3072
#define G4H  (4 * H_SZ)       // 12288
#define KTOT (H_SZ + E_SZ)    // 3840 : [h | x_t]
#define HXP  3848             // LDS row pitch in bf16 elems (16B aligned, conflict-free)

#if defined(__AMDGCN__) && __has_builtin(__builtin_amdgcn_global_load_async_to_lds_b128)
#define HAVE_ASYNC_COPY 1
#else
#define HAVE_ASYNC_COPY 0
#endif

__device__ __forceinline__ __bf16 f2bf(float f) {
    union { float f; unsigned u; } v; v.f = f;
    unsigned r = v.u + 0x7FFFu + ((v.u >> 16) & 1u);   // round to nearest even
    unsigned short h = (unsigned short)(r >> 16);
    return __builtin_bit_cast(__bf16, h);
}

__device__ __forceinline__ float sigm(float x) { return 1.0f / (1.0f + __expf(-x)); }

#if HAVE_ASYNC_COPY
// Direct memory->LDS copy, 16B per lane, tracked by ASYNCcnt (no VGPR round trip).
// AS1 ptr == flat ptr numerically; AS3 ptr == low 32 bits of flat LDS address.
__device__ __forceinline__ void async_b128(void* lds, const void* g) {
    __builtin_amdgcn_global_load_async_to_lds_b128(
        (__attribute__((address_space(1))) v4i*)(uintptr_t)g,
        (__attribute__((address_space(3))) v4i*)(unsigned)(uintptr_t)lds,
        0, 0);
}
__device__ __forceinline__ void wait_async0() {
#if __has_builtin(__builtin_amdgcn_s_wait_asynccnt)
    __builtin_amdgcn_s_wait_asynccnt(0);
#else
    asm volatile("s_wait_asynccnt 0" ::: "memory");
#endif
}
#endif

// A fragment (16x32 bf16, row-major source, leading dim lda):
// lanes 0-15 : M=lane,    K = {kb..kb+7,  kb+16..kb+23}
// lanes16-31 : M=lane-16, K = {kb+8..+15, kb+24..kb+31}
__device__ __forceinline__ v16bf load_a_frag(const __bf16* A, int lda,
                                             int mbase, int kbase, int lane) {
    int mr = lane & 15;
    int kq = (lane >> 4) << 3;                 // 0 or 8
    const __bf16* p = A + (long)(mbase + mr) * lda + kbase + kq;
    v16bf a;
    *(uint4*)&a       = *(const uint4*)p;        // K = kb+kq .. +7
    *((uint4*)&a + 1) = *(const uint4*)(p + 16); // K = kb+kq+16 .. +23
    return a;
}

// B fragment for gates = A * W^T : B[k][n] = W[n][k], W row-major [Nrows, ldw].
// lanes 0-15 : N=lane,    K = {kb..kb+7, kb+16..kb+23} ; lanes 16-31 : N=lane-16, K = {kb+8..+15, kb+24..kb+31}
__device__ __forceinline__ v16bf load_bT_frag(const __bf16* W, long ldw,
                                              int nbase, int kbase, int lane) {
    int nr = lane & 15;
    int kq = (lane >> 4) << 3;                 // 0 or 8
    const __bf16* p = W + (long)(nbase + nr) * ldw + kbase + kq;
    v16bf b;
    *(uint4*)&b       = *(const uint4*)p;       // K = kb+kq .. +7
    *((uint4*)&b + 1) = *(const uint4*)(p + 16); // K = kb+kq+16 .. +23
    return b;
}

__device__ __forceinline__ v8f wmma_bf16(v16bf a, v16bf b, v8f c) {
    return __builtin_amdgcn_wmma_f32_16x16x32_bf16(false, a, false, b,
                                                   (short)0, c, false, false);
}

// ---------------------------------------------------------------------------
// Prep kernels (run once per launch; all traffic trivial vs. the recurrence)
// ---------------------------------------------------------------------------
__global__ void k_pack_wcat(const float* __restrict__ Whh, const float* __restrict__ Wih,
                            __bf16* __restrict__ Wcat) {
    const long total = (long)G4H * KTOT;
    for (long i = (long)blockIdx.x * blockDim.x + threadIdx.x; i < total;
         i += (long)gridDim.x * blockDim.x) {
        long n = i / KTOT; int k = (int)(i - n * KTOT);
        float v = (k < H_SZ) ? Whh[n * H_SZ + k] : Wih[n * E_SZ + (k - H_SZ)];
        Wcat[i] = f2bf(v);
    }
}

__global__ void k_pack_wout(const float* __restrict__ W, __bf16* __restrict__ Wb) {
    const long total = (long)E_SZ * H_SZ;
    for (long i = (long)blockIdx.x * blockDim.x + threadIdx.x; i < total;
         i += (long)gridDim.x * blockDim.x)
        Wb[i] = f2bf(W[i]);
}

__global__ void k_bias(const float* __restrict__ bih, const float* __restrict__ bhh,
                       float* __restrict__ bias) {
    int i = blockIdx.x * blockDim.x + threadIdx.x;
    if (i < G4H) bias[i] = bih[i] + bhh[i];
}

// X_bf layout: row (t*32 + b), 768 bf16 cols  (x_t tile is contiguous)
__global__ void k_embed(const int* __restrict__ tok, const float* __restrict__ emb,
                        __bf16* __restrict__ X) {
    const long total = (long)B_SZ * T_SZ * E_SZ;
    for (long i = (long)blockIdx.x * blockDim.x + threadIdx.x; i < total;
         i += (long)gridDim.x * blockDim.x) {
        int e  = (int)(i % E_SZ);
        long bt = i / E_SZ;
        int m  = (int)(bt & 31);
        int t  = (int)(bt >> 5);
        int tk = tok[m * T_SZ + t];
        X[((long)t * B_SZ + m) * E_SZ + e] = f2bf(emb[(long)tk * E_SZ + e]);
    }
}

__global__ void k_init_state(const float* __restrict__ h0, const float* __restrict__ c0,
                             __bf16* __restrict__ hbf, float* __restrict__ c) {
    int i = blockIdx.x * blockDim.x + threadIdx.x;
    if (i < B_SZ * H_SZ) { c[i] = c0[i]; hbf[i] = f2bf(h0[i]); }
}

// ---------------------------------------------------------------------------
// Recurrent step: 96 blocks x 256 threads. Block owns hidden slice j0..j0+31,
// computes all 4 gates for it. [h|x_t] staged in LDS (32 x 3848 bf16) via
// GLOBAL_LOAD_ASYNC_TO_LDS (ASYNCcnt path) when available.
// wave w: gate g = w>>1, n-subtile nt = w&1; two 16x16 accumulators (M=0..31).
// ---------------------------------------------------------------------------
__global__ __launch_bounds__(256) void k_step(const __bf16* __restrict__ Wcat,
                                              const float* __restrict__ bias,
                                              const __bf16* __restrict__ Xbf,
                                              const __bf16* __restrict__ hin,
                                              __bf16* __restrict__ hout,
                                              float* __restrict__ c,
                                              __bf16* __restrict__ hs,
                                              int t) {
    extern __shared__ char smem[];
    __bf16* sHX = (__bf16*)smem;                                 // 32 x HXP bf16
    float*  sG  = (float*)(smem + (size_t)B_SZ * HXP * 2);       // 4 x 32 x 32 f32

    const int tid = threadIdx.x;

    // stage [h | x_t] (32 rows x 3840 bf16) -> LDS, pitch HXP
#if HAVE_ASYNC_COPY
    {
        const __bf16* xsrc = Xbf + (long)t * B_SZ * E_SZ;
        for (int ch = tid; ch < B_SZ * 384; ch += 256) {
            int row = ch / 384, off = ch % 384;
            async_b128(sHX + (long)row * HXP + off * 8, hin + (long)row * H_SZ + off * 8);
        }
        for (int ch = tid; ch < B_SZ * 96; ch += 256) {
            int row = ch / 96, off = ch % 96;
            async_b128(sHX + (long)row * HXP + H_SZ + off * 8,
                       xsrc + (long)row * E_SZ + off * 8);
        }
        wait_async0();
    }
#else
    {
        const uint4* src = (const uint4*)hin;                    // 8 bf16 per uint4
        for (int ch = tid; ch < B_SZ * 384; ch += 256) {
            int row = ch / 384, off = ch % 384;
            *(uint4*)(sHX + (long)row * HXP + off * 8) = src[(long)row * 384 + off];
        }
        const uint4* xs = (const uint4*)(Xbf + (long)t * B_SZ * E_SZ);
        for (int ch = tid; ch < B_SZ * 96; ch += 256) {
            int row = ch / 96, off = ch % 96;
            *(uint4*)(sHX + (long)row * HXP + H_SZ + off * 8) = xs[(long)row * 96 + off];
        }
    }
#endif
    __syncthreads();

    const int w    = tid >> 5;
    const int lane = tid & 31;
    const int g    = w >> 1;
    const int nt   = w & 1;
    const int j0   = blockIdx.x * 32;
    const int n0   = g * H_SZ + j0 + nt * 16;     // row block of packed W

    v8f acc0 = {}, acc1 = {};
    for (int k = 0; k < KTOT; k += 32) {
        v16bf a0 = load_a_frag(sHX, HXP, 0,  k, lane);
        v16bf a1 = load_a_frag(sHX, HXP, 16, k, lane);
        v16bf b  = load_bT_frag(Wcat, KTOT, n0, k, lane);
        acc0 = wmma_bf16(a0, b, acc0);
        acc1 = wmma_bf16(a1, b, acc1);
    }

    // deposit gate pre-activations (+bias) into LDS: sG[g][m][jc]
    const float bval = bias[n0 + (lane & 15)];
    const int col   = nt * 16 + (lane & 15);
    const int rbase = (lane >> 4) * 8;
#pragma unroll
    for (int r = 0; r < 8; ++r) {
        sG[((g * 32) + (rbase + r))      * 32 + col] = acc0[r] + bval;
        sG[((g * 32) + (16 + rbase + r)) * 32 + col] = acc1[r] + bval;
    }
    __syncthreads();

    // fused cell update for this 32(batch) x 32(hidden) slice
    for (int idx = tid; idx < B_SZ * 32; idx += 256) {
        int m = idx >> 5, jc = idx & 31;
        int j = j0 + jc;
        float iv = sigm(sG[(0 * 32 + m) * 32 + jc]);
        float fv = sigm(sG[(1 * 32 + m) * 32 + jc]);
        float gv = tanhf(sG[(2 * 32 + m) * 32 + jc]);
        float ov = sigm(sG[(3 * 32 + m) * 32 + jc]);
        long ci  = (long)m * H_SZ + j;
        float cv = fv * c[ci] + iv * gv;
        c[ci]    = cv;
        float hv = ov * tanhf(cv);
        __bf16 hb = f2bf(hv);
        hout[ci] = hb;                                         // next step's A
        hs[((long)m * T_SZ + t) * H_SZ + j] = hb;              // row b*T+t for proj
    }
}

// ---------------------------------------------------------------------------
// Output projection: out[16384,768] = hs[16384,3072](bf16) @ W_out^T + b_out
// grid (512, 6), 8 waves/block, wave tile = 32(M) x 16(N)
// ---------------------------------------------------------------------------
__global__ __launch_bounds__(256) void k_proj(const __bf16* __restrict__ hs,
                                              const __bf16* __restrict__ Wout,
                                              const float* __restrict__ bout,
                                              float* __restrict__ out) {
    const int tid   = threadIdx.x;
    const int w     = tid >> 5;
    const int lane  = tid & 31;
    const int mbase = blockIdx.x * 32;
    const int nbase = (blockIdx.y * 8 + w) * 16;

    v8f acc0 = {}, acc1 = {};
    for (int k = 0; k < H_SZ; k += 32) {
        v16bf a0 = load_a_frag(hs, H_SZ, mbase,      k, lane);
        v16bf a1 = load_a_frag(hs, H_SZ, mbase + 16, k, lane);
        v16bf b  = load_bT_frag(Wout, H_SZ, nbase,   k, lane);
        acc0 = wmma_bf16(a0, b, acc0);
        acc1 = wmma_bf16(a1, b, acc1);
    }
    const int col   = nbase + (lane & 15);
    const float bv  = bout[col];
    const int rbase = (lane >> 4) * 8;
#pragma unroll
    for (int r = 0; r < 8; ++r) {
        out[(long)(mbase + rbase + r)      * E_SZ + col] = acc0[r] + bv;
        out[(long)(mbase + 16 + rbase + r) * E_SZ + col] = acc1[r] + bv;
    }
}

// ---------------------------------------------------------------------------
// Host side
// ---------------------------------------------------------------------------
static inline size_t alignup(size_t x) { return (x + 255) & ~(size_t)255; }

extern "C" void kernel_launch(void* const* d_in, const int* in_sizes, int n_in,
                              void* d_out, int out_size, void* d_ws, size_t ws_size,
                              hipStream_t stream) {
    const int*   tokens = (const int*)  d_in[0];
    const float* emb    = (const float*)d_in[1];
    const float* Wih    = (const float*)d_in[2];
    const float* Whh    = (const float*)d_in[3];
    const float* bih    = (const float*)d_in[4];
    const float* bhh    = (const float*)d_in[5];
    const float* h0     = (const float*)d_in[6];
    const float* c0     = (const float*)d_in[7];
    const float* Wout   = (const float*)d_in[8];
    const float* bout   = (const float*)d_in[9];
    float* out = (float*)d_out;

    // workspace carve-up
    char* p = (char*)d_ws;
    size_t off = 0;
    __bf16* ws_Wcat = (__bf16*)(p + off); off = alignup(off + (size_t)G4H * KTOT * 2);
    __bf16* ws_Wout = (__bf16*)(p + off); off = alignup(off + (size_t)E_SZ * H_SZ * 2);
    __bf16* ws_X    = (__bf16*)(p + off); off = alignup(off + (size_t)B_SZ * T_SZ * E_SZ * 2);
    float*  ws_bias = (float*) (p + off); off = alignup(off + (size_t)G4H * 4);
    float*  ws_c    = (float*) (p + off); off = alignup(off + (size_t)B_SZ * H_SZ * 4);
    __bf16* ws_h0   = (__bf16*)(p + off); off = alignup(off + (size_t)B_SZ * H_SZ * 2);
    __bf16* ws_h1   = (__bf16*)(p + off); off = alignup(off + (size_t)B_SZ * H_SZ * 2);
    __bf16* ws_hs   = (__bf16*)(p + off); off = alignup(off + (size_t)B_SZ * T_SZ * H_SZ * 2);
    (void)ws_size; (void)in_sizes; (void)n_in; (void)out_size;

    // dynamic LDS for the step kernel: 32*3848 bf16 + 4*32*32 f32 = 262,656 B
    const int SMEM = (int)((size_t)B_SZ * HXP * 2 + 4 * 32 * 32 * 4);
    (void)hipFuncSetAttribute((const void*)k_step,
                              hipFuncAttributeMaxDynamicSharedMemorySize, SMEM);

    // prep
    k_pack_wcat <<<4096, 256, 0, stream>>>(Whh, Wih, ws_Wcat);
    k_pack_wout <<<512,  256, 0, stream>>>(Wout, ws_Wout);
    k_bias      <<<(G4H + 255) / 256, 256, 0, stream>>>(bih, bhh, ws_bias);
    k_embed     <<<2048, 256, 0, stream>>>(tokens, emb, ws_X);
    k_init_state<<<(B_SZ * H_SZ + 255) / 256, 256, 0, stream>>>(h0, c0, ws_h0, ws_c);

    // 512 sequential LSTM steps (stream ordering = recurrence dependency)
    for (int t = 0; t < T_SZ; ++t) {
        const __bf16* hin = (t & 1) ? ws_h1 : ws_h0;
        __bf16*      hout = (t & 1) ? ws_h0 : ws_h1;
        k_step<<<H_SZ / 32, 256, SMEM, stream>>>(ws_Wcat, ws_bias, ws_X,
                                                 hin, hout, ws_c, ws_hs, t);
    }

    // output projection
    dim3 pg((B_SZ * T_SZ) / 32, (E_SZ / 16) / 8);
    k_proj<<<pg, 256, 0, stream>>>(ws_hs, ws_Wout, bout, out);
}
